// Attention_9560597201123
// MI455X (gfx1250) — matrix operand order repaired and checked by reading, not compiled, not run on hardware
//
#include <hip/hip_runtime.h>
#include <math.h>

// ---------------------------------------------------------------------------
// Multi-head attention (B=4, N=2048, E=1024, H=16, D=64) on CDNA5 WMMA.
// All GEMM work runs on v_wmma_f32_16x16x32_f16 (f16 in, f32 accumulate).
// GEMMs: 64x128 block tiles, 32x32 per wave (4 WMMA / k-step / wave).
// Attention: flash-style streaming, K tile staged via async load-to-LDS.
// ---------------------------------------------------------------------------

typedef _Float16 half_t;
typedef __attribute__((ext_vector_type(16))) _Float16 v16h;
typedef __attribute__((ext_vector_type(8)))  _Float16 v8h;
typedef __attribute__((ext_vector_type(8)))  float    v8f;
typedef __attribute__((ext_vector_type(4)))  int      v4i;

#define EMB      1024
#define THREE_E  3072
#define SEQ      2048
#define BATCH    4
#define NHEAD    16
#define HDIM     64
#define ATT_SCALE 0.125f   // 1/sqrt(64)

// Optional CDNA5 async global->LDS path (ASYNCcnt-tracked), guarded so the
// kernel still builds with a manual-copy fallback on toolchains without it.
#if defined(__has_builtin)
# if __has_builtin(__builtin_amdgcn_global_load_async_to_lds_b128) && \
     __has_builtin(__builtin_amdgcn_s_wait_asynccnt)
#  define USE_ASYNC_LDS 1
# endif
#endif
#ifndef USE_ASYNC_LDS
# define USE_ASYNC_LDS 0
#endif

#if USE_ASYNC_LDS
typedef __attribute__((address_space(1))) v4i* global_v4i_ptr;
typedef __attribute__((address_space(3))) v4i* lds_v4i_ptr;
#endif

union V16H { v16h v; v8h h[2]; };

__device__ __forceinline__ v8f wmma16x16x32(v16h a, v16h b, v8f c) {
  // D = A(16x32) * B(32x16) + C, f32 accumulate
  return __builtin_amdgcn_wmma_f32_16x16x32_f16(
      /*neg_a=*/false, a, /*neg_b=*/false, b,
      /*c_mod=*/(short)0, c, /*reuse_a=*/false, /*reuse_b=*/false);
}

// A fragment (16x32 f16). Source is row-major [m][k], leading dim lda halves.
// Per ISA: lane<16 holds row m=lane, halves = k {0..7,16..23}; lane>=16 adds 8.
__device__ __forceinline__ v16h load_a_frag(const half_t* base, int lda, int kbase, int lane) {
  int m = lane & 15, hi = lane >> 4;
  const half_t* p = base + (size_t)m * lda + kbase + hi * 8;
  V16H u;
  u.h[0] = *(const v8h*)(p);        // k = kbase + hi*8 .. +7
  u.h[1] = *(const v8h*)(p + 16);   // k = kbase + 16 + hi*8 .. +7
  return u.v;
}

// A fragment from an f32 row-major source, converting to f16 on the fly.
__device__ __forceinline__ v16h load_a_frag_f32(const float* base, int lda, int kbase, int lane) {
  int m = lane & 15, hi = lane >> 4;
  const float* p = base + (size_t)m * lda + kbase + hi * 8;
  const float4* q0 = (const float4*)(p);
  const float4* q1 = (const float4*)(p + 16);
  float4 a0 = q0[0], a1 = q0[1], b0 = q1[0], b1 = q1[1];
  float t[16] = {a0.x,a0.y,a0.z,a0.w, a1.x,a1.y,a1.z,a1.w,
                 b0.x,b0.y,b0.z,b0.w, b1.x,b1.y,b1.z,b1.w};
  v16h r;
#pragma unroll
  for (int i = 0; i < 16; ++i) r[i] = (_Float16)t[i];
  return r;
}

// B fragment (32x16 f16). Source is "n-major": column n has contiguous k.
// Per ISA: lane holds column n=lane&15; halves = k {0..7,16..23}; lane>=16 adds 8.
__device__ __forceinline__ v16h load_b_frag(const half_t* base, int ldn, int kbase, int lane) {
  int n = lane & 15, hi = lane >> 4;
  const half_t* p = base + (size_t)n * ldn + kbase + hi * 8;
  V16H u;
  u.h[0] = *(const v8h*)(p);
  u.h[1] = *(const v8h*)(p + 16);
  return u.v;
}

// One 16-byte async global->LDS copy (no-op if builtin unavailable).
__device__ __forceinline__ void async_copy_b128(const half_t* g, half_t* l) {
#if USE_ASYNC_LDS
  __builtin_amdgcn_global_load_async_to_lds_b128(
      (global_v4i_ptr)(g), (lds_v4i_ptr)(l), /*offset=*/0, /*cpol=*/0);
#else
  (void)g; (void)l;
#endif
}

// ---------------------------------------------------------------------------
// Kernel 1: qkv = x @ qkv_w + qkv_b, written head-split as f16 q/k/v [B,H,N,D]
// Block = 256 threads (8 waves) -> 64(M) x 128(C) tile; wave = 32x32 (4 WMMA).
// ---------------------------------------------------------------------------
__global__ void __launch_bounds__(256)
qkv_wmma_kernel(const float* __restrict__ x, const float* __restrict__ w,
                const float* __restrict__ bias,
                half_t* __restrict__ qh, half_t* __restrict__ kh, half_t* __restrict__ vh) {
  __shared__ __align__(16) half_t Bt[128][40];  // weight tile, transposed [c][k]

  const int lane = threadIdx.x & 31;
  const int wave = threadIdx.x >> 5;
  const int wm = wave >> 2;                     // 0..1 : 32-row sub-tile
  const int wn = wave & 3;                      // 0..3 : 32-col sub-tile
  const int m0 = blockIdx.y * 64;
  const int c0 = blockIdx.x * 128;

  v8f acc[2][2] = {{{}, {}}, {{}, {}}};
  const float* arow0 = x + (size_t)(m0 + wm * 32)      * EMB;
  const float* arow1 = x + (size_t)(m0 + wm * 32 + 16) * EMB;

  for (int k0 = 0; k0 < EMB; k0 += 32) {
    __syncthreads();
    // Stage 32(k) x 128(c) f32 weights -> f16 LDS, transposed to n-major.
#pragma unroll
    for (int i = threadIdx.x; i < 32 * 128; i += 256) {
      int kk = i >> 7, cc = i & 127;
      Bt[cc][kk] = (_Float16)w[(size_t)(k0 + kk) * THREE_E + c0 + cc];
    }
    __syncthreads();
    v16h af0 = load_a_frag_f32(arow0, EMB, k0, lane);
    v16h af1 = load_a_frag_f32(arow1, EMB, k0, lane);
    v16h bf0 = load_b_frag(&Bt[wn * 32][0],      40, 0, lane);
    v16h bf1 = load_b_frag(&Bt[wn * 32 + 16][0], 40, 0, lane);
    acc[0][0] = wmma16x16x32(af0, bf0, acc[0][0]);
    acc[0][1] = wmma16x16x32(af0, bf1, acc[0][1]);
    acc[1][0] = wmma16x16x32(af1, bf0, acc[1][0]);
    acc[1][1] = wmma16x16x32(af1, bf1, acc[1][1]);
  }

  // Epilogue: add bias, scatter into q/k/v f16 [B,H,N,D].
  const int hi = lane >> 4;
  const int ncol = lane & 15;
#pragma unroll
  for (int ni = 0; ni < 2; ++ni) {
    const int c = c0 + wn * 32 + ni * 16 + ncol;
    const float bval = bias[c];
    const int which = c / EMB;
    const int e = c % EMB;
    const int hh = e / HDIM;
    const int dd = e & (HDIM - 1);
    half_t* dst = (which == 0) ? qh : (which == 1) ? kh : vh;
#pragma unroll
    for (int mi = 0; mi < 2; ++mi) {
#pragma unroll
      for (int r = 0; r < 8; ++r) {
        int row = m0 + wm * 32 + mi * 16 + r + hi * 8;   // 0..8191 over B*N
        int bb = row >> 11;                              // /2048
        int nn = row & (SEQ - 1);
        dst[(((size_t)bb * NHEAD + hh) * SEQ + nn) * HDIM + dd] =
            (_Float16)(acc[mi][ni][r] + bval);
      }
    }
  }
}

// ---------------------------------------------------------------------------
// Kernel 2: flash attention per (b,h). Block: 8 waves x 16 q-rows = 128 rows.
// Streams 32-key tiles; K staged once per block via async global->LDS copy.
// ---------------------------------------------------------------------------
__global__ void __launch_bounds__(256)
attn_wmma_kernel(const half_t* __restrict__ q, const half_t* __restrict__ k,
                 const half_t* __restrict__ v, half_t* __restrict__ outh) {
  __shared__ __align__(16) half_t Ks[32][72];       // K tile [key][d] (async-staged)
  __shared__ __align__(16) half_t Vt[64][40];       // V tile transposed [d][kk]
  __shared__ __align__(16) half_t P[8][16][40];     // per-wave probs [m][kk]

  const int lane = threadIdx.x & 31;
  const int wave = threadIdx.x >> 5;
  const int b = blockIdx.z, h = blockIdx.y;
  const int row0 = blockIdx.x * 128 + wave * 16;
  const int hi = lane >> 4;
  const int ncol = lane & 15;

  const half_t* qp = q + (((size_t)b * NHEAD + h) * SEQ) * HDIM;
  const half_t* kp = k + (((size_t)b * NHEAD + h) * SEQ) * HDIM;
  const half_t* vp = v + (((size_t)b * NHEAD + h) * SEQ) * HDIM;

  // Q fragments for this wave's 16 rows (d = 0..31 and 32..63), kept resident.
  const v16h qf0 = load_a_frag(qp + (size_t)row0 * HDIM, HDIM, 0,  lane);
  const v16h qf1 = load_a_frag(qp + (size_t)row0 * HDIM, HDIM, 32, lane);

  v8f o[4] = {{}, {}, {}, {}};                  // O accum, d-groups of 16
  float mi[8], li[8];
#pragma unroll
  for (int r = 0; r < 8; ++r) { mi[r] = -INFINITY; li[r] = 0.0f; }

  // Per-thread async-copy slot for the 32x64 K tile: one b128 each.
  const int krow = threadIdx.x >> 3;              // 0..31 (key)
  const int kch  = (threadIdx.x & 7) * 8;         // 0..56 (d, halves)

  for (int j = 0; j < SEQ; j += 32) {
    if (j + 32 < SEQ) {                           // hint next K/V tile toward L2
      __builtin_prefetch(kp + (size_t)(j + 32) * HDIM, 0, 0);
      __builtin_prefetch(vp + (size_t)(j + 32) * HDIM, 0, 0);
    }
    __syncthreads();

#if USE_ASYNC_LDS
    // K tile: 4KB byte-for-byte global->LDS, one async b128 per thread.
    async_copy_b128(kp + (size_t)(j + krow) * HDIM + kch, &Ks[krow][kch]);
#else
#pragma unroll
    for (int i = threadIdx.x; i < 32 * 64; i += 256) {
      int kk = i >> 6, dd = i & 63;
      Ks[kk][dd] = kp[(size_t)(j + kk) * HDIM + dd];
    }
#endif
    // V tile 32(kk) x 64(d) -> LDS transposed [d][kk] (needs VALU transpose).
#pragma unroll
    for (int i = threadIdx.x; i < 32 * 64; i += 256) {
      int kk = i >> 6, dd = i & 63;
      Vt[dd][kk] = vp[(size_t)(j + kk) * HDIM + dd];
    }
#if USE_ASYNC_LDS
    __builtin_amdgcn_s_wait_asynccnt(0);
#endif
    __syncthreads();

    // S = Q @ K^T for 32 keys (two 16-key column groups, two d chunks each).
    v8f s0 = {}, s1 = {};
    {
      v16h b00 = load_b_frag(&Ks[0][0],  72, 0,  lane);
      v16h b01 = load_b_frag(&Ks[0][0],  72, 32, lane);
      v16h b10 = load_b_frag(&Ks[16][0], 72, 0,  lane);
      v16h b11 = load_b_frag(&Ks[16][0], 72, 32, lane);
      s0 = wmma16x16x32(qf0, b00, s0);
      s0 = wmma16x16x32(qf1, b01, s0);
      s1 = wmma16x16x32(qf0, b10, s1);
      s1 = wmma16x16x32(qf1, b11, s1);
    }

    // Online softmax over the 32-key tile; row stats live per-lane (8 rows).
#pragma unroll
    for (int r = 0; r < 8; ++r) {
      float a0 = (float)s0[r] * ATT_SCALE;
      float a1 = (float)s1[r] * ATT_SCALE;
      float mx = fmaxf(a0, a1);
#pragma unroll
      for (int off = 1; off < 16; off <<= 1) mx = fmaxf(mx, __shfl_xor(mx, off, 32));
      float mnew  = fmaxf(mi[r], mx);
      float alpha = __expf(mi[r] - mnew);
      float p0 = __expf(a0 - mnew);
      float p1 = __expf(a1 - mnew);
      float ps = p0 + p1;
#pragma unroll
      for (int off = 1; off < 16; off <<= 1) ps += __shfl_xor(ps, off, 32);
      li[r] = li[r] * alpha + ps;
      mi[r] = mnew;
      o[0][r] *= alpha; o[1][r] *= alpha; o[2][r] *= alpha; o[3][r] *= alpha;
      P[wave][r + hi * 8][ncol]      = (_Float16)p0;
      P[wave][r + hi * 8][16 + ncol] = (_Float16)p1;
    }

    // O += P(16x32) @ V(32x64): one A-frag from LDS, four d-group WMMAs.
    v16h pf = load_a_frag(&P[wave][0][0], 40, 0, lane);
#pragma unroll
    for (int dg = 0; dg < 4; ++dg) {
      v16h vf = load_b_frag(&Vt[dg * 16][0], 40, 0, lane);
      o[dg] = wmma16x16x32(pf, vf, o[dg]);
    }
  }

  // Finalize: O /= l, write f16 attention output in [B, N, (H D)] layout.
#pragma unroll
  for (int r = 0; r < 8; ++r) {
    float inv = 1.0f / li[r];
    int row = row0 + r + hi * 8;
    size_t obase = ((size_t)b * SEQ + row) * EMB + (size_t)h * HDIM;
#pragma unroll
    for (int dg = 0; dg < 4; ++dg) {
      outh[obase + dg * 16 + ncol] = (_Float16)(o[dg][r] * inv);
    }
  }
}

// ---------------------------------------------------------------------------
// Kernel 3: out = attn_out @ proj_w + proj_b  (f16 A from ws, f32 output)
// Same 64x128 / 32x32-per-wave structure as kernel 1.
// ---------------------------------------------------------------------------
__global__ void __launch_bounds__(256)
proj_wmma_kernel(const half_t* __restrict__ a, const float* __restrict__ w,
                 const float* __restrict__ bias, float* __restrict__ out) {
  __shared__ __align__(16) half_t Bt[128][40];

  const int lane = threadIdx.x & 31;
  const int wave = threadIdx.x >> 5;
  const int wm = wave >> 2;
  const int wn = wave & 3;
  const int m0 = blockIdx.y * 64;
  const int c0 = blockIdx.x * 128;

  v8f acc[2][2] = {{{}, {}}, {{}, {}}};
  const half_t* arow0 = a + (size_t)(m0 + wm * 32)      * EMB;
  const half_t* arow1 = a + (size_t)(m0 + wm * 32 + 16) * EMB;

  for (int k0 = 0; k0 < EMB; k0 += 32) {
    __syncthreads();
#pragma unroll
    for (int i = threadIdx.x; i < 32 * 128; i += 256) {
      int kk = i >> 7, cc = i & 127;
      Bt[cc][kk] = (_Float16)w[(size_t)(k0 + kk) * EMB + c0 + cc];
    }
    __syncthreads();
    v16h af0 = load_a_frag(arow0, EMB, k0, lane);
    v16h af1 = load_a_frag(arow1, EMB, k0, lane);
    v16h bf0 = load_b_frag(&Bt[wn * 32][0],      40, 0, lane);
    v16h bf1 = load_b_frag(&Bt[wn * 32 + 16][0], 40, 0, lane);
    acc[0][0] = wmma16x16x32(af0, bf0, acc[0][0]);
    acc[0][1] = wmma16x16x32(af0, bf1, acc[0][1]);
    acc[1][0] = wmma16x16x32(af1, bf0, acc[1][0]);
    acc[1][1] = wmma16x16x32(af1, bf1, acc[1][1]);
  }

  const int hi = lane >> 4;
  const int ncol = lane & 15;
#pragma unroll
  for (int ni = 0; ni < 2; ++ni) {
    const int c = c0 + wn * 32 + ni * 16 + ncol;
    const float bval = bias[c];
#pragma unroll
    for (int mi = 0; mi < 2; ++mi) {
#pragma unroll
      for (int r = 0; r < 8; ++r) {
        int row = m0 + wm * 32 + mi * 16 + r + hi * 8;
        out[(size_t)row * EMB + c] = acc[mi][ni][r] + bval;
      }
    }
  }
}

// ---------------------------------------------------------------------------
extern "C" void kernel_launch(void* const* d_in, const int* in_sizes, int n_in,
                              void* d_out, int out_size, void* d_ws, size_t ws_size,
                              hipStream_t stream) {
  (void)in_sizes; (void)n_in; (void)out_size; (void)ws_size;
  const float* x      = (const float*)d_in[0];
  const float* qkv_w  = (const float*)d_in[1];
  const float* qkv_b  = (const float*)d_in[2];
  const float* proj_w = (const float*)d_in[3];
  const float* proj_b = (const float*)d_in[4];
  float* out = (float*)d_out;

  // Workspace carve-up (f16): q, k, v in [B,H,N,D] + attn output [B,N,E].
  const size_t HSZ = (size_t)BATCH * NHEAD * SEQ * HDIM;   // 8,388,608 halves
  half_t* qh  = (half_t*)d_ws;
  half_t* kh  = qh + HSZ;
  half_t* vh  = kh + HSZ;
  half_t* aoh = vh + HSZ;                                   // 64 MB total

  qkv_wmma_kernel<<<dim3(THREE_E / 128, (BATCH * SEQ) / 64), 256, 0, stream>>>(
      x, qkv_w, qkv_b, qh, kh, vh);
  attn_wmma_kernel<<<dim3(SEQ / 128, NHEAD, BATCH), 256, 0, stream>>>(
      qh, kh, vh, aoh);
  proj_wmma_kernel<<<dim3(EMB / 128, (BATCH * SEQ) / 64), 256, 0, stream>>>(
      aoh, proj_w, proj_b, out);
}
